// WeightedReadOutPair_7112465842767
// MI455X (gfx1250) — matrix operand order repaired and checked by reading, not compiled, not run on hardware
//
#include <hip/hip_runtime.h>

// ---------------------------------------------------------------------------
// Types for CDNA5 WMMA (wave32, 16x16x32 bf16 -> f32)
// ---------------------------------------------------------------------------
typedef __attribute__((ext_vector_type(16))) __bf16 v16bf;
typedef __attribute__((ext_vector_type(8)))  float  v8f;

union BF16x16 {
    v16bf          v;
    __bf16         b[16];
    unsigned short s[16];
};

// Fast reciprocal (v_rcp_f32) with safe fallback
__device__ __forceinline__ float fast_rcp(float x) {
#if __has_builtin(__builtin_amdgcn_rcpf)
    return __builtin_amdgcn_rcpf(x);
#else
    return 1.0f / x;
#endif
}

// sigmoid via v_tanh_f32 (gfx1250 trans op) when available:
//   sigma(x) = 0.5*tanh(0.5x) + 0.5   -> mul + tanh + fma
__device__ __forceinline__ float sigmoid_f(float x) {
#if __has_builtin(__builtin_amdgcn_tanhf)
    return fmaf(0.5f, __builtin_amdgcn_tanhf(0.5f * x), 0.5f);
#else
    return fast_rcp(1.0f + __expf(-x));
#endif
}

__device__ __forceinline__ float silu_f(float x) { return x * sigmoid_f(x); }

// ---------------------------------------------------------------------------
// Kernel 1: y2 = 0.5*(X @ W1) + 0.5*b1   (bias folded: h_pair = y2[i]+y2[j])
// One thread per output element (65536 threads).
// ---------------------------------------------------------------------------
__global__ void __launch_bounds__(256) k_precompute(
    const float* __restrict__ x,   // [1024][64]
    const float* __restrict__ W,   // [64][64] row-major (k, f)
    const float* __restrict__ b,   // [64]
    float* __restrict__ y)         // [1024][64]
{
    const int t = blockIdx.x * blockDim.x + threadIdx.x;
    const int n = t >> 6;
    const int f = t & 63;
    const float* xr = x + n * 64;
    float s = 0.0f;
#pragma unroll
    for (int k = 0; k < 64; ++k)
        s = fmaf(xr[k], W[k * 64 + f], s);
    y[t] = 0.5f * s + 0.5f * b[f];
}

// ---------------------------------------------------------------------------
// Kernel 2: pack 64x64 f32 weights (row-major [k][n]) into bf16 WMMA B-frag
// layout. Fragment (nf,kf) = 32 lanes x 16 bf16, contiguous per lane so the
// main kernel loads each half as one b128.
// Lane L of frag: N = nf*16 + (L%16);  K = kf*32 + (L/16)*16 + e, e=0..15.
// ---------------------------------------------------------------------------
__global__ void __launch_bounds__(256) k_pack(
    const float* __restrict__ W,       // [64][64]
    unsigned short* __restrict__ P)    // [8 frags][32 lanes][16]
{
    const int t    = blockIdx.x * blockDim.x + threadIdx.x;
    const int e    = t & 15;
    const int lane = (t >> 4) & 31;
    const int frag = t >> 9;           // nf*2 + kf
    const int kf   = frag & 1;
    const int nf   = frag >> 1;
    const int n    = (nf << 4) | (lane & 15);
    const int k    = kf * 32 + (lane >> 4) * 16 + e;
    union { __bf16 b; unsigned short s; } cv;
    cv.b = (__bf16)W[k * 64 + n];
    P[t] = cv.s;
}

// ---------------------------------------------------------------------------
// Per-branch evaluation for one 16-pair tile (one wave):
//   a1 = silu(y2[i] + y2[j])                 (built directly into A-frag regs)
//   z2 = a1 @ W2  via 8x v_wmma_f32_16x16x32_bf16
//   res[v] = sum_n silu(z2 + b2[n]) * w3[n] + b3   (per-pair scalar)
// After the shfl reduction every lane holds res[] for M = v + 8*(lane/16).
// ---------------------------------------------------------------------------
__device__ __forceinline__ void branch_eval(
    const float* __restrict__ y,           // [1024][64] = 0.5*X@W1 + 0.5*b1
    const unsigned short* __restrict__ pB, // packed W2 (8 frags)
    const float* __restrict__ b2,          // [64]
    const float* __restrict__ w3,          // [64] (W3 is [64][1])
    float b3, int i, int jrow, int lane,
    float res[8])
{
    const int lm = lane & 15;
    const int hi = lane >> 4;
    const float* yi = y + i * 64;
    const float* yj = y + jrow * 64;

    // ---- build A fragments (16x32 bf16 each), kf = 0,1 -> K 0..31 / 32..63
    BF16x16 afr[2];
#pragma unroll
    for (int kf = 0; kf < 2; ++kf) {
#pragma unroll
        for (int c = 0; c < 2; ++c) {               // two 8-wide K chunks
            const int k0 = kf * 32 + hi * 16 + c * 8;
            float4 ai0 = *reinterpret_cast<const float4*>(yi + k0);
            float4 ai1 = *reinterpret_cast<const float4*>(yi + k0 + 4);
            float4 aj0 = *reinterpret_cast<const float4*>(yj + k0);
            float4 aj1 = *reinterpret_cast<const float4*>(yj + k0 + 4);
            float h[8];
            h[0] = ai0.x + aj0.x;
            h[1] = ai0.y + aj0.y;
            h[2] = ai0.z + aj0.z;
            h[3] = ai0.w + aj0.w;
            h[4] = ai1.x + aj1.x;
            h[5] = ai1.y + aj1.y;
            h[6] = ai1.z + aj1.z;
            h[7] = ai1.w + aj1.w;
#pragma unroll
            for (int e = 0; e < 8; ++e)
                afr[kf].b[c * 8 + e] = (__bf16)silu_f(h[e]);
        }
    }

    // ---- layer-2 WMMA over 4 N-tiles, fold layer-3 immediately
    float partial[8];
#pragma unroll
    for (int v = 0; v < 8; ++v) partial[v] = 0.0f;

#pragma unroll
    for (int nf = 0; nf < 4; ++nf) {
        v8f acc = {0.f, 0.f, 0.f, 0.f, 0.f, 0.f, 0.f, 0.f};
#pragma unroll
        for (int kf = 0; kf < 2; ++kf) {
            BF16x16 bfr;
            const unsigned short* p = pB + ((((nf << 1) | kf) * 32) + lane) * 16;
            *reinterpret_cast<uint4*>(&bfr.s[0]) = *reinterpret_cast<const uint4*>(p);
            *reinterpret_cast<uint4*>(&bfr.s[8]) = *reinterpret_cast<const uint4*>(p + 8);
            acc = __builtin_amdgcn_wmma_f32_16x16x32_bf16(
                /*neg_a=*/false, afr[kf].v, /*neg_b=*/false, bfr.v,
                /*c_mod=*/(short)0, acc, /*reuse_a=*/false, /*reuse_b=*/false);
        }
        const int   n  = (nf << 4) | lm;     // D layout: lane holds column n
        const float bn = b2[n];
        const float wn = w3[n];
#pragma unroll
        for (int v = 0; v < 8; ++v) {        // D layout: VGPR v -> M = v + 8*hi
            float z = acc[v] + bn;
            partial[v] = fmaf(silu_f(z), wn, partial[v]);
        }
    }

    // ---- reduce over the 16 lanes holding columns n of the same pair-row M
#pragma unroll
    for (int v = 0; v < 8; ++v) {
        float s = partial[v];
        s += __shfl_xor(s, 1);
        s += __shfl_xor(s, 2);
        s += __shfl_xor(s, 4);
        s += __shfl_xor(s, 8);
        res[v] = s + b3;
    }
}

// ---------------------------------------------------------------------------
// Main kernel: one wave per 16-pair tile (fixed i, j = j0..j0+15).
// 65536 waves total.
// ---------------------------------------------------------------------------
__global__ void __launch_bounds__(256) k_main(
    const float* __restrict__ yl, const float* __restrict__ yg,
    const unsigned short* __restrict__ Pl, const unsigned short* __restrict__ Pg,
    const float* __restrict__ lb2, const float* __restrict__ lw3,
    const float* __restrict__ lb3,
    const float* __restrict__ gb2, const float* __restrict__ gw3,
    const float* __restrict__ gb3,
    float* __restrict__ out)
{
    const int wave = (int)((blockIdx.x * blockDim.x + threadIdx.x) >> 5);
    const int lane = (int)(threadIdx.x & 31);
    const int lm   = lane & 15;
    const int hi   = lane >> 4;
    const int i    = wave >> 6;           // 0..1023
    const int j0   = (wave & 63) << 4;    // 0..1008 step 16
    const int jrow = j0 + lm;             // A-frag row M = lm -> pair (i, jrow)

    float m[8], g[8];
    branch_eval(yl, Pl, lb2, lw3, lb3[0], i, jrow, lane, m);
    branch_eval(yg, Pg, gb2, gw3, gb3[0], i, jrow, lane, g);

    if (lm == 0) {
#pragma unroll
        for (int v = 0; v < 8; ++v) {
            out[i * 1024 + j0 + hi * 8 + v] = m[v] * sigmoid_f(g[v]);
        }
    }
}

// ---------------------------------------------------------------------------
// Launch
// ---------------------------------------------------------------------------
extern "C" void kernel_launch(void* const* d_in, const int* in_sizes, int n_in,
                              void* d_out, int out_size, void* d_ws, size_t ws_size,
                              hipStream_t stream) {
    (void)in_sizes; (void)n_in; (void)out_size; (void)ws_size;

    const float* x   = (const float*)d_in[0];
    const float* lW1 = (const float*)d_in[1];
    const float* lb1 = (const float*)d_in[2];
    const float* lW2 = (const float*)d_in[3];
    const float* lb2 = (const float*)d_in[4];
    const float* lW3 = (const float*)d_in[5];
    const float* lb3 = (const float*)d_in[6];
    const float* gW1 = (const float*)d_in[7];
    const float* gb1 = (const float*)d_in[8];
    const float* gW2 = (const float*)d_in[9];
    const float* gb2 = (const float*)d_in[10];
    const float* gW3 = (const float*)d_in[11];
    const float* gb3 = (const float*)d_in[12];

    // workspace layout
    const size_t YBYTES = 1024 * 64 * sizeof(float);   // 256 KB each
    char* ws = (char*)d_ws;
    float*          yl = (float*)(ws);
    float*          yg = (float*)(ws + YBYTES);
    unsigned short* Pl = (unsigned short*)(ws + 2 * YBYTES);
    unsigned short* Pg = Pl + 4096;

    // Phase 1: hoisted first layers y = 0.5*(X@W1) + 0.5*b1 (bias folded;
    // per-pair pre-activation is then just y[i]+y[j])
    k_precompute<<<256, 256, 0, stream>>>(x, lW1, lb1, yl);
    k_precompute<<<256, 256, 0, stream>>>(x, gW1, gb1, yg);

    // Phase 2: pack layer-2 weights into bf16 WMMA B-fragment layout
    k_pack<<<16, 256, 0, stream>>>(lW2, Pl);
    k_pack<<<16, 256, 0, stream>>>(gW2, Pg);

    // Phase 3: per-pair gated MLP; one wave per 16-pair tile
    // 1024 i * 64 j-tiles = 65536 waves -> 8192 blocks of 256 (8 waves)
    k_main<<<8192, 256, 0, stream>>>(yl, yg, Pl, Pg,
                                     lb2, lW3, lb3,
                                     gb2, gW3, gb3,
                                     (float*)d_out);
}
